// GAT_42975442763994
// MI455X (gfx1250) — matrix operand order repaired and checked by reading, not compiled, not run on hardware
//
#include <hip/hip_runtime.h>
#include <hip/hip_fp16.h>

typedef __attribute__((ext_vector_type(16))) _Float16 v16h;
typedef __attribute__((ext_vector_type(8)))  _Float16 v8h;
typedef __attribute__((ext_vector_type(8)))  float    v8f;
typedef __attribute__((ext_vector_type(4)))  unsigned int v4u;
typedef __attribute__((ext_vector_type(8)))  int v8i;
typedef __attribute__((ext_vector_type(4)))  int v4i;

#define NEG_SLOPE 0.2f
#define GAT_EPS 1e-16f

#if __has_builtin(__builtin_amdgcn_tensor_load_to_lds)
#define HAVE_TDM 1
#endif

// ---------------- elementwise helpers ----------------

__global__ void k_f32_to_f16(const float* __restrict__ in, _Float16* __restrict__ out, int n) {
  int i = blockIdx.x * blockDim.x + threadIdx.x;
  if (i < n) out[i] = (_Float16)in[i];
}

__global__ void k_fill(float* __restrict__ p, float v, int n) {
  int i = blockIdx.x * blockDim.x + threadIdx.x;
  if (i < n) p[i] = v;
}

// h[i] += bias[i % F]; optional ReLU
__global__ void k_bias_act(float* __restrict__ h, const float* __restrict__ b,
                           int n, int F, int relu) {
  int i = blockIdx.x * blockDim.x + threadIdx.x;
  if (i >= n) return;
  float v = h[i] + b[i & (F - 1)];   // F is a power of two (128 / 64)
  if (relu) v = fmaxf(v, 0.f);
  h[i] = v;
}

// ---------------- WMMA GEMM: C[M x NCOL] = A[M x K] * B[K x NCOL] ----------------
// A,B row-major f16; C row-major f32.
// One block per 16-row M tile; wave w computes the 16x16 tile at column w*16.
//  - A tile (16 x K, contiguous 8/4 KB) staged to LDS by the Tensor Data Mover
//    (tensor_load_to_lds + s_wait_tensorcnt), fallback: async-to-LDS copy.
//  - B panel staged row-major via global_load_async_to_lds_b128 (coalesced),
//    then transposed LDS->LDS so B fragments are contiguous ds_load_b128.
// Fragment layouts per CDNA5 ISA 7.12.2.
template<int K, int NCOL>
__global__ void k_wmma_gemm(const _Float16* __restrict__ A, const _Float16* __restrict__ B,
                            float* __restrict__ C, int M) {
  constexpr int NTILES = NCOL / 16;
  constexpr int NTHR   = NTILES * 32;
  __shared__ _Float16 sA [16 * K];       // A tile, row-major [16][K]
  __shared__ _Float16 sB [K * NCOL];     // B panel, row-major staging
  __shared__ _Float16 sBT[NCOL * K];     // B transposed [n][k]

  const int tid   = threadIdx.x;
  const int tileM = blockIdx.x;
  const int wave  = tid >> 5;            // N tile index
  const int lane  = tid & 31;
  const int row   = lane & 15;
  const int hi    = lane >> 4;

  // ---- stage A tile into LDS ----
#ifdef HAVE_TDM
  if (wave == 0) {
    const unsigned total = 16u * K;      // elements (2B each)
    unsigned long long ga = (unsigned long long)(uintptr_t)(A + (size_t)tileM * 16 * K);
    unsigned lds = (unsigned)(uintptr_t)sA;
    // D# group 0: count=1 | lds_addr | global_addr[56:0] | type=2
    v4u g0 = { 1u, lds, (unsigned)(ga & 0xffffffffu),
               (unsigned)((ga >> 32) & 0x01ffffffu) | (2u << 30) };
    // D# group 1: data_size=2B; tensor_dim0 = tile_dim0 = total; dim0 stride = total
    v8i g1 = { (int)(1u << 16),
               (int)((total & 0xffffu) << 16),
               (int)(((total >> 16) & 0xffffu) | (1u << 16)),
               (int)(total << 16),
               0,
               (int)total,
               0, 0 };
    v4i gz = { 0, 0, 0, 0 };
#if __clang_major__ >= 23
    v8i gz8 = { 0, 0, 0, 0, 0, 0, 0, 0 };
    __builtin_amdgcn_tensor_load_to_lds(g0, g1, gz, gz, gz8, 0);
#else
    __builtin_amdgcn_tensor_load_to_lds(g0, g1, gz, gz, 0);
#endif
    __builtin_amdgcn_s_wait_tensorcnt(0);
  }
#else
  {
    const _Float16* Abase = A + (size_t)tileM * 16 * K;
    const unsigned ldsBase = (unsigned)(uintptr_t)sA;
    for (int c = tid; c < (16 * K) / 8; c += NTHR) {
      unsigned loff = ldsBase + (unsigned)c * 16u;
      unsigned long long ga = (unsigned long long)(uintptr_t)(Abase + c * 8);
      asm volatile("global_load_async_to_lds_b128 %0, %1, off"
                   :: "v"(loff), "v"(ga) : "memory");
    }
  }
#endif

  // ---- async-copy B panel row-major into LDS (coalesced 16B chunks) ----
  {
    const unsigned ldsBase = (unsigned)(uintptr_t)sB;
    for (int c = tid; c < (K * NCOL) / 8; c += NTHR) {
      unsigned loff = ldsBase + (unsigned)c * 16u;
      unsigned long long ga = (unsigned long long)(uintptr_t)(B + (size_t)c * 8);
      asm volatile("global_load_async_to_lds_b128 %0, %1, off"
                   :: "v"(loff), "v"(ga) : "memory");
    }
  }
  asm volatile("s_wait_asynccnt 0x0" ::: "memory");
  __syncthreads();

  // ---- transpose B panel in LDS: sBT[n*K + k] = sB[k*NCOL + n] ----
  for (int c = tid; c < (K * NCOL) / 8; c += NTHR) {
    int k  = c / (NCOL / 8);
    int n0 = (c - k * (NCOL / 8)) * 8;
    v8h v = *(const v8h*)(sB + (size_t)k * NCOL + n0);
    #pragma unroll
    for (int j = 0; j < 8; ++j) sBT[(size_t)(n0 + j) * K + k] = v[j];
  }
  __syncthreads();

  v8f acc = {};
  #pragma unroll
  for (int k0 = 0; k0 < K; k0 += 32) {
    // A fragment: lane holds A[row][k0+hi*8 .. +7] then A[row][k0+16+hi*8 .. +7]
    v16h a;
    {
      const _Float16* ap = sA + row * K + k0 + hi * 8;
      v8h lo = *(const v8h*)ap;
      v8h hv = *(const v8h*)(ap + 16);
      #pragma unroll
      for (int i = 0; i < 8; ++i) { a[i] = lo[i]; a[8 + i] = hv[i]; }
    }
    // B fragment: lane holds column (wave*16+row), K = k0+hi*8 .. +7 then k0+16+hi*8 .. +7
    v16h b;
    {
      const _Float16* bp = sBT + (size_t)(wave * 16 + row) * K + k0 + hi * 8;
      v8h b0 = *(const v8h*)bp;
      v8h b1 = *(const v8h*)(bp + 16);
      #pragma unroll
      for (int i = 0; i < 8; ++i) { b[i] = b0[i]; b[8 + i] = b1[i]; }
    }
    acc = __builtin_amdgcn_wmma_f32_16x16x32_f16(
        /*neg_a=*/false, a, /*neg_b=*/false, b,
        /*c_mod=*/(short)0, acc, /*reuse_a=*/false, /*reuse_b=*/false);
  }

  #pragma unroll
  for (int r = 0; r < 8; ++r) {
    int m = tileM * 16 + hi * 8 + r;
    if (m < M) C[(size_t)m * NCOL + wave * 16 + row] = acc[r];
  }
}

// ---------------- edge-phase kernels (wave-per-edge, L2-resident gathers) ----------------

__device__ inline float waveReduceAdd(float v) {
  #pragma unroll
  for (int off = 16; off > 0; off >>= 1) v += __shfl_down(v, off, 32);
  return v;
}

// float atomic max via signed/unsigned ordering trick (init must be -inf)
__device__ inline void atomicMaxF(float* addr, float val) {
  if (val >= 0.f) atomicMax((int*)addr, __float_as_int(val));
  else            atomicMin((unsigned int*)addr, __float_as_uint(val));
}

// e[w] = att . leakyrelu(xl[src] + xr[dst]); m[dst] = max(m[dst], e[w])
__global__ void k_edge_logits(const float* __restrict__ xl, const float* __restrict__ xr,
                              const int* __restrict__ src, const int* __restrict__ dst,
                              const float* __restrict__ att, float* __restrict__ elog,
                              float* __restrict__ m, int E, int F) {
  int w = blockIdx.x * (blockDim.x >> 5) + (threadIdx.x >> 5);
  if (w >= E) return;
  int lane = threadIdx.x & 31;
  int s = src[w], d = dst[w];
  const float* pl = xl + (size_t)s * F;
  const float* pr = xr + (size_t)d * F;
  float sum = 0.f;
  for (int i = lane; i < F; i += 32) {
    float z = pl[i] + pr[i];
    float lr = z > 0.f ? z : z * NEG_SLOPE;
    sum += lr * att[i];
  }
  sum = waveReduceAdd(sum);
  if (lane == 0) { elog[w] = sum; atomicMaxF(m + d, sum); }
}

// a[w] = exp(e[w] - m[dst]); s[dst] += a[w]   (a overwrites e in place)
__global__ void k_edge_exp(const int* __restrict__ dst, const float* __restrict__ m,
                           float* __restrict__ elog, float* __restrict__ ssum, int E) {
  int i = blockIdx.x * blockDim.x + threadIdx.x;
  if (i >= E) return;
  int d = dst[i];
  float a = __expf(elog[i] - m[d]);
  elog[i] = a;
  atomicAdd(ssum + d, a);
}

// out[dst] += (a[w] / (s[dst]+eps)) * xl[src]
__global__ void k_edge_aggregate(const float* __restrict__ xl, const int* __restrict__ src,
                                 const int* __restrict__ dst, const float* __restrict__ a,
                                 const float* __restrict__ ssum, float* __restrict__ out,
                                 int E, int F) {
  int w = blockIdx.x * (blockDim.x >> 5) + (threadIdx.x >> 5);
  if (w >= E) return;
  int lane = threadIdx.x & 31;
  int s = src[w], d = dst[w];
  const float* pl = xl + (size_t)s * F;
  __builtin_prefetch((const void*)(pl + lane), 0, 3);   // global_prefetch_b8
  float alpha = a[w] / (ssum[d] + GAT_EPS);
  float* po = out + (size_t)d * F;
  for (int i = lane; i < F; i += 32)
    atomicAdd(po + i, alpha * pl[i]);
}

// ---------------- launcher ----------------

extern "C" void kernel_launch(void* const* d_in, const int* in_sizes, int n_in,
                              void* d_out, int out_size, void* d_ws, size_t ws_size,
                              hipStream_t stream) {
  const float* x    = (const float*)d_in[0];
  const int*   ei   = (const int*)  d_in[1];
  const float* Wl1  = (const float*)d_in[2];
  const float* Wr1  = (const float*)d_in[3];
  const float* att1 = (const float*)d_in[4];
  const float* b1   = (const float*)d_in[5];
  const float* Wl2  = (const float*)d_in[6];
  const float* Wr2  = (const float*)d_in[7];
  const float* att2 = (const float*)d_in[8];
  const float* b2   = (const float*)d_in[9];

  const int FIN = 256, H = 128, C = 64;
  const int N = in_sizes[0] / FIN;   // 50000 (multiple of 16)
  const int E = in_sizes[1] / 2;     // 1.6M
  const int* srcIdx = ei;
  const int* dstIdx = ei + E;

  // carve workspace (256B aligned slices)
  char* wsp = (char*)d_ws;
  auto carve = [&](size_t bytes) -> char* {
    char* p = wsp; wsp += (bytes + 255) & ~(size_t)255; return p;
  };
  _Float16* xh    = (_Float16*)carve((size_t)N * FIN * sizeof(_Float16));
  _Float16* Wl1h  = (_Float16*)carve((size_t)FIN * H * sizeof(_Float16));
  _Float16* Wr1h  = (_Float16*)carve((size_t)FIN * H * sizeof(_Float16));
  float*    xl1   = (float*)   carve((size_t)N * H * sizeof(float));
  float*    xr1   = (float*)   carve((size_t)N * H * sizeof(float));
  float*    ebuf  = (float*)   carve((size_t)E * sizeof(float));
  float*    mbuf  = (float*)   carve((size_t)N * sizeof(float));
  float*    sbuf  = (float*)   carve((size_t)N * sizeof(float));
  float*    h1    = (float*)   carve((size_t)N * H * sizeof(float));
  _Float16* h1h   = (_Float16*)carve((size_t)N * H * sizeof(_Float16));
  _Float16* Wl2h  = (_Float16*)carve((size_t)H * C * sizeof(_Float16));
  _Float16* Wr2h  = (_Float16*)carve((size_t)H * C * sizeof(_Float16));
  float*    xl2   = (float*)   carve((size_t)N * C * sizeof(float));
  float*    xr2   = (float*)   carve((size_t)N * C * sizeof(float));
  float*    out2  = (float*)d_out;

  const int TB = 256;
  auto grid1 = [&](int n) { return dim3((n + TB - 1) / TB); };
  const int wavesPerBlk = TB / 32;
  dim3 egrid((E + wavesPerBlk - 1) / wavesPerBlk);
  const float NEG_INF = -__builtin_huge_valf();

  // ---------------- layer 1 ----------------
  k_f32_to_f16<<<grid1(N * FIN), TB, 0, stream>>>(x, xh, N * FIN);
  k_f32_to_f16<<<grid1(FIN * H), TB, 0, stream>>>(Wl1, Wl1h, FIN * H);
  k_f32_to_f16<<<grid1(FIN * H), TB, 0, stream>>>(Wr1, Wr1h, FIN * H);

  dim3 ggrid1((N + 15) / 16);
  k_wmma_gemm<256, 128><<<ggrid1, (128 / 16) * 32, 0, stream>>>(xh, Wl1h, xl1, N);
  k_wmma_gemm<256, 128><<<ggrid1, (128 / 16) * 32, 0, stream>>>(xh, Wr1h, xr1, N);

  k_fill<<<grid1(N), TB, 0, stream>>>(mbuf, NEG_INF, N);
  k_fill<<<grid1(N), TB, 0, stream>>>(sbuf, 0.f, N);
  k_fill<<<grid1(N * H), TB, 0, stream>>>(h1, 0.f, N * H);

  k_edge_logits<<<egrid, TB, 0, stream>>>(xl1, xr1, srcIdx, dstIdx, att1, ebuf, mbuf, E, H);
  k_edge_exp<<<grid1(E), TB, 0, stream>>>(dstIdx, mbuf, ebuf, sbuf, E);
  k_edge_aggregate<<<egrid, TB, 0, stream>>>(xl1, srcIdx, dstIdx, ebuf, sbuf, h1, E, H);
  k_bias_act<<<grid1(N * H), TB, 0, stream>>>(h1, b1, N * H, H, /*relu=*/1);

  // ---------------- layer 2 ----------------
  k_f32_to_f16<<<grid1(N * H), TB, 0, stream>>>(h1, h1h, N * H);
  k_f32_to_f16<<<grid1(H * C), TB, 0, stream>>>(Wl2, Wl2h, H * C);
  k_f32_to_f16<<<grid1(H * C), TB, 0, stream>>>(Wr2, Wr2h, H * C);

  dim3 ggrid2((N + 15) / 16);
  k_wmma_gemm<128, 64><<<ggrid2, (64 / 16) * 32, 0, stream>>>(h1h, Wl2h, xl2, N);
  k_wmma_gemm<128, 64><<<ggrid2, (64 / 16) * 32, 0, stream>>>(h1h, Wr2h, xr2, N);

  k_fill<<<grid1(N), TB, 0, stream>>>(mbuf, NEG_INF, N);
  k_fill<<<grid1(N), TB, 0, stream>>>(sbuf, 0.f, N);
  k_fill<<<grid1(N * C), TB, 0, stream>>>(out2, 0.f, N * C);

  k_edge_logits<<<egrid, TB, 0, stream>>>(xl2, xr2, srcIdx, dstIdx, att2, ebuf, mbuf, E, C);
  k_edge_exp<<<grid1(E), TB, 0, stream>>>(dstIdx, mbuf, ebuf, sbuf, E);
  k_edge_aggregate<<<egrid, TB, 0, stream>>>(xl2, srcIdx, dstIdx, ebuf, sbuf, out2, E, C);
  k_bias_act<<<grid1(N * C), TB, 0, stream>>>(out2, b2, N * C, C, /*relu=*/0);
}
